// VQVAEEncoder_48172353192564
// MI455X (gfx1250) — compile-verified, matrix operand order checked by reading, not run on hardware
//
#include <hip/hip_runtime.h>

// ---------------------------------------------------------------------------
// VQ-VAE encoder for MI455X (gfx1250, wave32, WMMA).
// Heavy GEMMs (conv layers 2-6, VQ distance) run on v_wmma_f32_16x16x32_bf16.
// Conv: 64(o)x128(t) block tile, 4 accumulators/wave, double-buffered LDS.
// ---------------------------------------------------------------------------

typedef unsigned short ushort_t;
typedef __attribute__((ext_vector_type(16))) __bf16 v16bf;
typedef __attribute__((ext_vector_type(8)))  float  v8f;

__device__ __forceinline__ ushort_t f2bf(float f) {
  unsigned int u = __builtin_bit_cast(unsigned int, f);
  u += 0x7FFFu + ((u >> 16) & 1u);              // round-to-nearest-even
  return (ushort_t)(u >> 16);
}

// ---------------------------------------------------------------------------
// Layer 1: IN=12 -> D=256, k=4, pad=2  (tiny K=48 reduction, fp32 VALU)
// x: (B, 4096, 12)  ->  y: (B, 256, 4097)
// ---------------------------------------------------------------------------
__global__ void conv1_kernel(const float* __restrict__ x,
                             const float* __restrict__ w1,
                             const float* __restrict__ b1,
                             float* __restrict__ y) {
  __shared__ float wsh[48];
  __shared__ float bsh;
  const int o = blockIdx.y;
  const int b = blockIdx.z;
  if (threadIdx.x < 48) wsh[threadIdx.x] = w1[o * 48 + threadIdx.x];
  if (threadIdx.x == 0) bsh = b1[o];
  __syncthreads();
  const int t = blockIdx.x * blockDim.x + threadIdx.x;
  if (t >= 4097) return;
  float acc = bsh;
#pragma unroll
  for (int j = 0; j < 4; ++j) {
    const int tin = t + j - 2;
    if (tin >= 0 && tin < 4096) {
      const float* xp = x + ((size_t)b * 4096 + tin) * 12;
#pragma unroll
      for (int c = 0; c < 12; ++c) acc += wsh[c * 4 + j] * xp[c];
    }
  }
  y[((size_t)b * 256 + o) * 4097 + t] = acc;
}

// ---------------------------------------------------------------------------
// Per-channel sum / sum-of-squares over (B, T)
// ---------------------------------------------------------------------------
__global__ void stats_kernel(const float* __restrict__ y, int T,
                             float* __restrict__ stats) {
  const int o = blockIdx.x, b = blockIdx.y;
  const float* p = y + ((size_t)b * 256 + o) * (size_t)T;
  float s = 0.f, q = 0.f;
  for (int t = threadIdx.x; t < T; t += 256) { float v = p[t]; s += v; q += v * v; }
  __shared__ float sh[512];
  sh[threadIdx.x] = s; sh[256 + threadIdx.x] = q;
  __syncthreads();
  for (int w = 128; w > 0; w >>= 1) {
    if (threadIdx.x < w) {
      sh[threadIdx.x]       += sh[threadIdx.x + w];
      sh[256 + threadIdx.x] += sh[256 + threadIdx.x + w];
    }
    __syncthreads();
  }
  if (threadIdx.x == 0) {
    atomicAdd(&stats[o],       sh[0]);
    atomicAdd(&stats[256 + o], sh[256]);
  }
}

// Fold BN into per-channel (scale, shift) for the consumer conv
__global__ void bn_finalize_kernel(const float* __restrict__ stats,
                                   const float* __restrict__ g,
                                   const float* __restrict__ be,
                                   float N, float* __restrict__ sc,
                                   float* __restrict__ tr) {
  const int o = threadIdx.x;
  const float m = stats[o] / N;
  const float v = stats[256 + o] / N - m * m;
  const float s = g[o] * rsqrtf(v + 1e-5f);
  sc[o] = s;
  tr[o] = be[o] - m * s;
}

// ---------------------------------------------------------------------------
// Activation prep: y fp32 -> bf16 act = relu(s*y + t)  (B, 256, T)
// Hoists BN+ReLU+convert out of the conv hot loop; halves conv read bytes.
// ---------------------------------------------------------------------------
__global__ void actprep_kernel(const float* __restrict__ y,
                               const float* __restrict__ sc,
                               const float* __restrict__ tr,
                               ushort_t* __restrict__ act, int T) {
  const int c = blockIdx.y, b = blockIdx.z;
  const int t = blockIdx.x * 256 + threadIdx.x;
  if (t >= T) return;
  const size_t idx = ((size_t)b * 256 + c) * (size_t)T + t;
  float f = y[idx] * sc[c] + tr[c];
  act[idx] = f2bf(f > 0.f ? f : 0.f);
}

// ---------------------------------------------------------------------------
// Weight prep: w (256, 256, 4) fp32 -> bf16 in CDNA5 16-bit A-fragment layout:
// wperm[o_tile(16)][kchunk(32)][lane(32)][e(16)]
//   lane = (o&15) + 16*half, half=(kk>>3)&1, e=(kk&7)+((kk&16)?8:0), kk=cj&31
// ---------------------------------------------------------------------------
__global__ void wprep_kernel(const float* __restrict__ w,
                             ushort_t* __restrict__ wperm) {
  const int idx = blockIdx.x * 256 + threadIdx.x;   // 0..262143
  const int o = idx >> 10, cj = idx & 1023;
  const int c = cj >> 2, j = cj & 3;
  const ushort_t v = f2bf(w[(o * 256 + c) * 4 + j]);
  const int kchunk = cj >> 5, kk = cj & 31;
  const int half = (kk >> 3) & 1;
  const int e = (kk & 7) + ((kk & 16) ? 8 : 0);
  const int lane = (o & 15) + 16 * half;
  wperm[((((o >> 4) * 32 + kchunk) * 32) + lane) * 16 + e] = v;
}

// ---------------------------------------------------------------------------
// Conv layers 2..6 as GEMM: Y[o][t] = sum_{cj} W[o][cj] * act[cj][t]
// cj = c*4+j (K=1024). Block = 8 waves covering 64(o) x 128(t);
// wave w: t-tile = w, 4 accumulators over the block's 4 o-tiles sharing one
// B fragment. LDS B staging is double-buffered: one barrier per K-chunk.
// ---------------------------------------------------------------------------
__global__ void __launch_bounds__(256)
conv_wmma_kernel(const ushort_t* __restrict__ wperm,
                 const ushort_t* __restrict__ act,
                 const float* __restrict__ bias,
                 float* __restrict__ yout,
                 int T_in, int T_out, int pad, int transposed) {
  __shared__ __attribute__((aligned(32))) ushort_t Bbuf0[8][32][16];
  __shared__ __attribute__((aligned(32))) ushort_t Bbuf1[8][32][16];
  const int tid = threadIdx.x;
  const int wave = tid >> 5, lane = tid & 31;
  const int tBlock = blockIdx.x * 128;
  const int oBase4 = blockIdx.y * 4;           // first global o-tile of this block
  const int b = blockIdx.z;

  const ushort_t* actb = act + (size_t)b * 256 * (size_t)T_in;

  v8f acc[4] = {{}, {}, {}, {}};

  // stage 32(kk) x 128(t) bf16 tile, writer-side permuted into fragments
  auto stage = [&](int chunk, ushort_t (*buf)[32][16]) {
#pragma unroll
    for (int i = 0; i < 16; ++i) {
      const int idx = tid + i * 256;
      const int kk = idx >> 7, tloc = idx & 127;
      const int cj = chunk * 32 + kk;
      const int c = cj >> 2, j = cj & 3;
      const int tg = tBlock + tloc;
      const int tin = tg + j - pad;
      ushort_t v = 0;
      if (tg < T_out && tin >= 0 && tin < T_in)
        v = actb[(size_t)c * (size_t)T_in + tin];
      const int half = (kk >> 3) & 1;
      const int e = (kk & 7) + ((kk & 16) ? 8 : 0);
      buf[tloc >> 4][(tloc & 15) + 16 * half][e] = v;
    }
  };
  // 4 WMMAs sharing one B fragment (4 o-tiles)
  auto compute = [&](int chunk, const ushort_t (*buf)[32][16]) {
    const v16bf bm = *(const v16bf*)&buf[wave][lane][0];
#pragma unroll
    for (int i = 0; i < 4; ++i) {
      const int ot = oBase4 + i;
      const v16bf a = *(const v16bf*)&wperm[(((ot * 32 + chunk) * 32) + lane) * 16];
      acc[i] = __builtin_amdgcn_wmma_f32_16x16x32_bf16(false, a, false, bm,
                                                       (short)0, acc[i], false, false);
    }
  };

  stage(0, Bbuf0);
  __syncthreads();
  for (int chunk = 0; chunk < 32; chunk += 2) {
    stage(chunk + 1, Bbuf1);
    compute(chunk, Bbuf0);
    __syncthreads();
    if (chunk + 2 < 32) stage(chunk + 2, Bbuf0);
    compute(chunk + 1, Bbuf1);
    __syncthreads();
  }

  const int half = lane >> 4, n = lane & 15;
  const int t = tBlock + wave * 16 + n;
  if (t < T_out) {
#pragma unroll
    for (int i = 0; i < 4; ++i) {
#pragma unroll
      for (int r = 0; r < 8; ++r) {
        const int o = (oBase4 + i) * 16 + r + 8 * half;
        const float v = acc[i][r] + bias[o];
        if (transposed) yout[((size_t)b * 4096 + t) * 256 + o] = v;       // z_e (B,T,D)
        else            yout[((size_t)b * 256 + o) * (size_t)T_out + t] = v;
      }
    }
  }
}

// ---------------------------------------------------------------------------
// Codebook prep: bf16 B-fragment layout + per-row squared norms
// cbperm[k_tile(64)][chunk(8)][lane(32)][e(16)]
// ---------------------------------------------------------------------------
__global__ void cbprep_kernel(const float* __restrict__ cb,
                              ushort_t* __restrict__ cbperm,
                              float* __restrict__ cnorm) {
  const int k = blockIdx.x, d = threadIdx.x;
  const float v = cb[k * 256 + d];
  const int chunk = d >> 5, kd = d & 31;
  const int half = (kd >> 3) & 1;
  const int e = (kd & 7) + ((kd & 16) ? 8 : 0);
  const int lane = (k & 15) + 16 * half;
  cbperm[((((k >> 4) * 8 + chunk) * 32) + lane) * 16 + e] = f2bf(v);
  __shared__ float sh[256];
  sh[d] = v * v;
  __syncthreads();
  for (int w = 128; w > 0; w >>= 1) { if (d < w) sh[d] += sh[d + w]; __syncthreads(); }
  if (d == 0) cnorm[k] = sh[0];
}

// ---------------------------------------------------------------------------
// VQ argmin: per wave, 16 z-rows resident as 8 A-fragments; sweep 64 k-tiles
// (8 WMMAs each over D=256), argmin of (|c|^2 - 2 z.c); shfl_xor reduction.
// ---------------------------------------------------------------------------
__global__ void __launch_bounds__(256)
vq_argmin_kernel(const float* __restrict__ ze,
                 const ushort_t* __restrict__ cbperm,
                 const float* __restrict__ cnorm,
                 int* __restrict__ ids) {
  const int tid = threadIdx.x;
  const int wave = tid >> 5, lane = tid & 31;
  const int half = lane >> 4, m = lane & 15;
  const int rowBase = blockIdx.x * 128 + wave * 16;
  const float* zp = ze + (size_t)(rowBase + m) * 256;

  v16bf areg[8];
#pragma unroll
  for (int chunk = 0; chunk < 8; ++chunk) {
    v16bf a;
#pragma unroll
    for (int e = 0; e < 16; e += 2) {
      const int d = chunk * 32 + e + 8 * half + ((e & 8) ? 8 : 0);
      const float2 f = *(const float2*)&zp[d];
      a[e]     = __builtin_bit_cast(__bf16, f2bf(f.x));
      a[e + 1] = __builtin_bit_cast(__bf16, f2bf(f.y));
    }
    areg[chunk] = a;
  }

  float minv[8]; int mini[8];
#pragma unroll
  for (int r = 0; r < 8; ++r) { minv[r] = 3.4e38f; mini[r] = 0; }

  for (int kt = 0; kt < 64; ++kt) {
    v8f acc = {};
#pragma unroll
    for (int chunk = 0; chunk < 8; ++chunk) {
      const v16bf bm = *(const v16bf*)&cbperm[(((kt * 8 + chunk) * 32) + lane) * 16];
      acc = __builtin_amdgcn_wmma_f32_16x16x32_bf16(false, areg[chunk], false, bm,
                                                    (short)0, acc, false, false);
    }
    const int kcol = kt * 16 + m;
    const float cn = cnorm[kcol];
#pragma unroll
    for (int r = 0; r < 8; ++r) {
      const float dd = cn - 2.f * acc[r];
      if (dd < minv[r]) { minv[r] = dd; mini[r] = kcol; }
    }
  }

#pragma unroll
  for (int r = 0; r < 8; ++r) {
#pragma unroll
    for (int msk = 8; msk >= 1; msk >>= 1) {
      const float ov = __shfl_xor(minv[r], msk, 32);
      const int   oi = __shfl_xor(mini[r], msk, 32);
      if (ov < minv[r] || (ov == minv[r] && oi < mini[r])) { minv[r] = ov; mini[r] = oi; }
    }
  }
  if (m == 0) {
#pragma unroll
    for (int r = 0; r < 8; ++r) ids[rowBase + r + 8 * half] = mini[r];
  }
}

// ---------------------------------------------------------------------------
// z_q gather + EMA accumulators (segment sums / counts via atomics)
// ---------------------------------------------------------------------------
__global__ void scatter_kernel(const float* __restrict__ ze,
                               const int* __restrict__ ids,
                               const float* __restrict__ cb,
                               float* __restrict__ zq,
                               float* __restrict__ sumcur,
                               float* __restrict__ counts) {
  const int n = blockIdx.x;
  const int d = threadIdx.x;
  const int id = ids[n];
  zq[(size_t)n * 256 + d] = cb[id * 256 + d];
  atomicAdd(&sumcur[id * 256 + d], ze[(size_t)n * 256 + d]);
  if (d == 0) atomicAdd(&counts[id], 1.0f);
}

__global__ void ecs_kernel(const float* __restrict__ ema_cs,
                           const float* __restrict__ counts,
                           float* __restrict__ ecs_out,
                           float* __restrict__ smoothed) {
  const int k = threadIdx.x;
  const float ne = ema_cs[k] * 0.99f + 0.01f * counts[k];
  ecs_out[k] = ne;
  __shared__ float sh[1024];
  sh[k] = ne;
  __syncthreads();
  for (int w = 512; w > 0; w >>= 1) { if (k < w) sh[k] += sh[k + w]; __syncthreads(); }
  const float n = sh[0];
  smoothed[k] = (ne + 1e-5f) / (n + 1024.f * 1e-5f) * n;
}

__global__ void newcb_kernel(const float* __restrict__ ema_w,
                             const float* __restrict__ sumcur,
                             const float* __restrict__ smoothed,
                             float* __restrict__ out) {
  const int idx = blockIdx.x * 256 + threadIdx.x;
  out[idx] = (ema_w[idx] * 0.99f + 0.01f * sumcur[idx]) / smoothed[idx >> 8];
}

// ---------------------------------------------------------------------------
extern "C" void kernel_launch(void* const* d_in, const int* in_sizes, int n_in,
                              void* d_out, int out_size, void* d_ws, size_t ws_size,
                              hipStream_t stream) {
  (void)in_sizes; (void)n_in; (void)out_size; (void)ws_size;
  const float* x = (const float*)d_in[0];
  const float* w[7]; const float* bb[7];
  for (int i = 1; i <= 6; ++i) { w[i] = (const float*)d_in[2 * i - 1]; bb[i] = (const float*)d_in[2 * i]; }
  const float* g[6]; const float* be[6];
  for (int i = 1; i <= 5; ++i) { g[i] = (const float*)d_in[11 + 2 * i]; be[i] = (const float*)d_in[12 + 2 * i]; }
  const float* cb     = (const float*)d_in[23];
  const float* ema_w  = (const float*)d_in[24];
  const float* ema_cs = (const float*)d_in[25];

  float* z_e  = (float*)d_out;                 // (32,4096,256)
  float* z_q  = z_e + 33554432;                // (32,4096,256)
  float* ncb  = z_e + 67108864;                // (1024,256)
  float* necs = z_e + 67371008;                // (1024,)

  char* ws = (char*)d_ws;
  size_t off = 0;
  auto bump = [&](size_t bytes) {
    char* p = ws + off;
    off = (off + bytes + 255) & ~(size_t)255;
    return p;
  };
  float*    yA       = (float*)bump((size_t)32 * 256 * 4097 * 4);
  float*    yB       = (float*)bump((size_t)32 * 256 * 4097 * 4);
  ushort_t* act      = (ushort_t*)bump((size_t)32 * 256 * 4097 * 2);
  float*    stats    = (float*)bump(2048);
  float*    sc       = (float*)bump(1024);
  float*    tr       = (float*)bump(1024);
  ushort_t* wperm    = (ushort_t*)bump(524288);
  ushort_t* cbperm   = (ushort_t*)bump(524288);
  float*    cnorm    = (float*)bump(4096);
  int*      ids      = (int*)bump(524288);
  float*    counts   = (float*)bump(4096);
  float*    sumcur   = (float*)bump(1048576);
  float*    smoothed = (float*)bump(4096);

  // ---- layer 1 (T: 4096 -> 4097, pad 2) ----
  conv1_kernel<<<dim3(17, 256, 32), 256, 0, stream>>>(x, w[1], bb[1], yA);
  hipMemsetAsync(stats, 0, 2048, stream);
  stats_kernel<<<dim3(256, 32), 256, 0, stream>>>(yA, 4097, stats);
  bn_finalize_kernel<<<1, 256, 0, stream>>>(stats, g[1], be[1], 32.f * 4097.f, sc, tr);

  // ---- layer 2 (4097 -> 4096, pad 1) ----
  actprep_kernel<<<dim3(17, 256, 32), 256, 0, stream>>>(yA, sc, tr, act, 4097);
  wprep_kernel<<<1024, 256, 0, stream>>>(w[2], wperm);
  conv_wmma_kernel<<<dim3(32, 4, 32), 256, 0, stream>>>(wperm, act, bb[2], yB, 4097, 4096, 1, 0);
  hipMemsetAsync(stats, 0, 2048, stream);
  stats_kernel<<<dim3(256, 32), 256, 0, stream>>>(yB, 4096, stats);
  bn_finalize_kernel<<<1, 256, 0, stream>>>(stats, g[2], be[2], 32.f * 4096.f, sc, tr);

  // ---- layer 3 (4096 -> 4097, pad 2) ----
  actprep_kernel<<<dim3(16, 256, 32), 256, 0, stream>>>(yB, sc, tr, act, 4096);
  wprep_kernel<<<1024, 256, 0, stream>>>(w[3], wperm);
  conv_wmma_kernel<<<dim3(33, 4, 32), 256, 0, stream>>>(wperm, act, bb[3], yA, 4096, 4097, 2, 0);
  hipMemsetAsync(stats, 0, 2048, stream);
  stats_kernel<<<dim3(256, 32), 256, 0, stream>>>(yA, 4097, stats);
  bn_finalize_kernel<<<1, 256, 0, stream>>>(stats, g[3], be[3], 32.f * 4097.f, sc, tr);

  // ---- layer 4 (4097 -> 4096, pad 1) ----
  actprep_kernel<<<dim3(17, 256, 32), 256, 0, stream>>>(yA, sc, tr, act, 4097);
  wprep_kernel<<<1024, 256, 0, stream>>>(w[4], wperm);
  conv_wmma_kernel<<<dim3(32, 4, 32), 256, 0, stream>>>(wperm, act, bb[4], yB, 4097, 4096, 1, 0);
  hipMemsetAsync(stats, 0, 2048, stream);
  stats_kernel<<<dim3(256, 32), 256, 0, stream>>>(yB, 4096, stats);
  bn_finalize_kernel<<<1, 256, 0, stream>>>(stats, g[4], be[4], 32.f * 4096.f, sc, tr);

  // ---- layer 5 (4096 -> 4097, pad 2) ----
  actprep_kernel<<<dim3(16, 256, 32), 256, 0, stream>>>(yB, sc, tr, act, 4096);
  wprep_kernel<<<1024, 256, 0, stream>>>(w[5], wperm);
  conv_wmma_kernel<<<dim3(33, 4, 32), 256, 0, stream>>>(wperm, act, bb[5], yA, 4096, 4097, 2, 0);
  hipMemsetAsync(stats, 0, 2048, stream);
  stats_kernel<<<dim3(256, 32), 256, 0, stream>>>(yA, 4097, stats);
  bn_finalize_kernel<<<1, 256, 0, stream>>>(stats, g[5], be[5], 32.f * 4097.f, sc, tr);

  // ---- layer 6 (4097 -> 4096, pad 1), stores z_e transposed (B,T,D) ----
  actprep_kernel<<<dim3(17, 256, 32), 256, 0, stream>>>(yA, sc, tr, act, 4097);
  wprep_kernel<<<1024, 256, 0, stream>>>(w[6], wperm);
  conv_wmma_kernel<<<dim3(32, 4, 32), 256, 0, stream>>>(wperm, act, bb[6], z_e, 4097, 4096, 1, 1);

  // ---- VQ ----
  cbprep_kernel<<<1024, 256, 0, stream>>>(cb, cbperm, cnorm);
  vq_argmin_kernel<<<1024, 256, 0, stream>>>(z_e, cbperm, cnorm, ids);

  hipMemsetAsync(counts, 0, 4096, stream);
  hipMemsetAsync(sumcur, 0, 1048576, stream);
  scatter_kernel<<<131072, 256, 0, stream>>>(z_e, ids, cb, z_q, sumcur, counts);

  ecs_kernel<<<1, 1024, 0, stream>>>(ema_cs, counts, necs, smoothed);
  newcb_kernel<<<1024, 256, 0, stream>>>(ema_w, sumcur, smoothed, ncb);
}
